// MOE_MLP_31731218383227
// MI455X (gfx1250) — matrix operand order repaired and checked by reading, not compiled, not run on hardware
//
#include <hip/hip_runtime.h>

// ---- problem constants (match reference) ----
#define BB 8
#define CC 1024
#define NNs 64
#define PPs 32
#define EE 3
#define HHd 4096
#define OOd 1024
#define AA (BB * NNs * PPs)   // 16384 tokens
#define NPP (NNs * PPs)       // 2048 tokens per batch image

typedef _Float16 v16h __attribute__((ext_vector_type(16)));
typedef _Float16 v8h  __attribute__((ext_vector_type(8)));
typedef float    v8f  __attribute__((ext_vector_type(8)));

static __device__ __forceinline__ v16h cat8(v8h lo, v8h hi) {
  return __builtin_shufflevector(lo, hi, 0,1,2,3,4,5,6,7,8,9,10,11,12,13,14,15);
}

// A fragment: 16x32 f16 tile at (row0, k). A row-major, leading dim = ld (== Ktot).
// ISA layout: lane<16 -> row=lane, K = [k..k+7] and [k+16..k+23];
//             lane>=16 -> row=lane-16, K = [k+8..k+15] and [k+24..k+31].
static __device__ __forceinline__ v16h load_afrag(const _Float16* __restrict__ A,
                                                  int ld, int row0, int k,
                                                  int r, int sub) {
  const _Float16* p = A + (size_t)(row0 + r) * ld + k + sub * 8;
  v8h lo = *(const v8h*)p;
  v8h hi = *(const v8h*)(p + 16);
  return cat8(lo, hi);
}

// B fragment: 32x16 f16 tile at (col0, k). B stored TRANSPOSED [N][K], ld == Ktot.
// ISA layout: col = lane%16, K = [k+8*(lane/16)..+7] and [k+16+8*(lane/16)..+7].
static __device__ __forceinline__ v16h load_bfrag(const _Float16* __restrict__ Bt,
                                                  int ld, int col0, int k,
                                                  int r, int sub) {
  const _Float16* p = Bt + (size_t)(col0 + r) * ld + k + sub * 8;
  v8h lo = *(const v8h*)p;
  v8h hi = *(const v8h*)(p + 16);
  return cat8(lo, hi);
}

// Main K loop: wave computes a 64x64 tile as 4x4 WMMA accumulators.
template <int KTOT>
static __device__ __forceinline__ void wmma_main_loop(
    const _Float16* __restrict__ A, const _Float16* __restrict__ Bt,
    int mBase, int nBase, int r, int sub, v8f acc[4][4]) {
  for (int k = 0; k < KTOT; k += 32) {
    v16h aF[4], bF[4];
#pragma unroll
    for (int mi = 0; mi < 4; ++mi)
      aF[mi] = load_afrag(A, KTOT, mBase + mi * 16, k, r, sub);
#pragma unroll
    for (int ni = 0; ni < 4; ++ni)
      bF[ni] = load_bfrag(Bt, KTOT, nBase + ni * 16, k, r, sub);
#pragma unroll
    for (int mi = 0; mi < 4; ++mi) {
#pragma unroll
      for (int ni = 0; ni < 4; ++ni) {
        acc[mi][ni] = __builtin_amdgcn_wmma_f32_16x16x32_f16(
            false, aF[mi], false, bF[ni], (short)0, acc[mi][ni], false, false);
      }
    }
  }
}

// ---------------- router: fp32 noisy top-2 gates ----------------
__global__ __launch_bounds__(256) void router_kernel(
    const float* __restrict__ x, const float* __restrict__ Wg,
    const float* __restrict__ bg, const float* __restrict__ Wn,
    const float* __restrict__ bn, const float* __restrict__ nu,
    float* __restrict__ gates) {
  int a = blockIdx.x * blockDim.x + threadIdx.x;   // token id
  int b = a >> 11;                                 // a / 2048
  int rem = a & (NPP - 1);
  const float* xp = x + (size_t)b * CC * NPP + rem;
  float g0 = 0.f, g1 = 0.f, g2 = 0.f, n0 = 0.f, n1 = 0.f, n2 = 0.f;
  for (int c = 0; c < CC; ++c) {
    float xv = xp[(size_t)c * NPP];                // coalesced across lanes (p)
    g0 = fmaf(xv, Wg[c * 3 + 0], g0);
    g1 = fmaf(xv, Wg[c * 3 + 1], g1);
    g2 = fmaf(xv, Wg[c * 3 + 2], g2);
    n0 = fmaf(xv, Wn[c * 3 + 0], n0);
    n1 = fmaf(xv, Wn[c * 3 + 1], n1);
    n2 = fmaf(xv, Wn[c * 3 + 2], n2);
  }
  n0 += bn[0]; n1 += bn[1]; n2 += bn[2];
  // softmax of noise logits
  float nm = fmaxf(n0, fmaxf(n1, n2));
  float e0 = __expf(n0 - nm) , e1 = __expf(n1 - nm), e2 = __expf(n2 - nm);
  // use precise expf for router (discrete decision downstream)
  e0 = expf(n0 - nm); e1 = expf(n1 - nm); e2 = expf(n2 - nm);
  float es = e0 + e1 + e2;
  float l[3];
  l[0] = g0 + bg[0] + nu[(size_t)a * 3 + 0] * (e0 / es);
  l[1] = g1 + bg[1] + nu[(size_t)a * 3 + 1] * (e1 / es);
  l[2] = g2 + bg[2] + nu[(size_t)a * 3 + 2] * (e2 / es);
  // exclude argmin; ties exclude the LARGER index (top_k keeps earlier indices)
  int ex = 0;
  if (l[1] <= l[ex]) ex = 1;
  if (l[2] <= l[ex]) ex = 2;
  float m = -3.4e38f;
  for (int e = 0; e < 3; ++e) if (e != ex) m = fmaxf(m, l[e]);
  float w[3]; float s = 0.f;
  for (int e = 0; e < 3; ++e) {
    w[e] = (e == ex) ? 0.f : expf(l[e] - m);
    s += w[e];
  }
  for (int e = 0; e < 3; ++e) gates[(size_t)a * 3 + e] = w[e] / s;
}

// ---------------- pack x: [B,C,N,P] f32 -> xh [A,C] f16 (LDS tile transpose) ----
__global__ __launch_bounds__(256) void pack_x_kernel(const float* __restrict__ x,
                                                     _Float16* __restrict__ xh) {
  __shared__ float tile[32][33];
  int a0 = blockIdx.x * 32;   // token tile (aligned: never crosses batch)
  int c0 = blockIdx.y * 32;
  int b = a0 >> 11;
  int rem0 = a0 & (NPP - 1);
  int tx = threadIdx.x, ty = threadIdx.y;
  for (int i = ty; i < 32; i += 8)
    tile[i][tx] = x[((size_t)(b * CC + c0 + i)) * NPP + rem0 + tx];  // [ci][ai]
  __syncthreads();
  for (int i = ty; i < 32; i += 8)
    xh[(size_t)(a0 + i) * CC + c0 + tx] = (_Float16)tile[tx][i];
}

// ---------------- weight transpose-convert: in [z][R][S] f32 -> out [z][S][R] f16
__global__ __launch_bounds__(256) void transpose_f32_f16(const float* __restrict__ in,
                                                         _Float16* __restrict__ out,
                                                         int R, int S) {
  __shared__ float tile[32][33];
  const float* inb = in + (size_t)blockIdx.z * R * S;
  _Float16* outb = out + (size_t)blockIdx.z * R * S;
  int s0 = blockIdx.x * 32, r0 = blockIdx.y * 32;
  int tx = threadIdx.x, ty = threadIdx.y;
  for (int i = ty; i < 32; i += 8)
    tile[i][tx] = inb[(size_t)(r0 + i) * S + s0 + tx];   // [ri][si]
  __syncthreads();
  for (int i = ty; i < 32; i += 8)
    outb[(size_t)(s0 + i) * R + r0 + tx] = (_Float16)tile[tx][i];
}

// ---------------- GEMM1: hh = gelu(xh @ W1[e] + b1[e]), f16 out ----------------
__global__ __launch_bounds__(256) void gemm1_gelu(const _Float16* __restrict__ Xh,
                                                  const _Float16* __restrict__ W1t,
                                                  const float* __restrict__ b1,
                                                  _Float16* __restrict__ Hout) {
  int lane = threadIdx.x & 31, wid = threadIdx.x >> 5;
  int r = lane & 15, sub = lane >> 4;
  int mBase = blockIdx.x * 128 + (wid >> 2) * 64;
  int nBase = blockIdx.y * 256 + (wid & 3) * 64;
  v8f acc[4][4] = {};
  wmma_main_loop<CC>(Xh, W1t, mBase, nBase, r, sub, acc);
#pragma unroll
  for (int mi = 0; mi < 4; ++mi) {
#pragma unroll
    for (int ni = 0; ni < 4; ++ni) {
      int n = nBase + ni * 16 + r;
      float bias = b1[n];
#pragma unroll
      for (int q = 0; q < 8; ++q) {
        int m = mBase + mi * 16 + sub * 8 + q;
        float v = acc[mi][ni][q] + bias;
        v = 0.5f * v * (1.0f + erff(v * 0.70710678118654752f));  // exact GELU
        Hout[(size_t)m * HHd + n] = (_Float16)v;
      }
    }
  }
}

// ---------------- GEMM2: tmp (+)= gate[:,e] * (hh @ W2[e] + b2[e]) ------------
__global__ __launch_bounds__(256) void gemm2_gate(const _Float16* __restrict__ Hh,
                                                  const _Float16* __restrict__ W2t,
                                                  const float* __restrict__ b2,
                                                  const float* __restrict__ gates,
                                                  int e, int accumulate,
                                                  float* __restrict__ outp) {
  int lane = threadIdx.x & 31, wid = threadIdx.x >> 5;
  int r = lane & 15, sub = lane >> 4;
  int mBase = blockIdx.x * 128 + (wid >> 2) * 64;
  int nBase = blockIdx.y * 256 + (wid & 3) * 64;
  v8f acc[4][4] = {};
  wmma_main_loop<HHd>(Hh, W2t, mBase, nBase, r, sub, acc);
#pragma unroll
  for (int mi = 0; mi < 4; ++mi) {
    float gv[8];
#pragma unroll
    for (int q = 0; q < 8; ++q)
      gv[q] = gates[(size_t)(mBase + mi * 16 + sub * 8 + q) * EE + e];
#pragma unroll
    for (int ni = 0; ni < 4; ++ni) {
      int n = nBase + ni * 16 + r;
      float bias = b2[n];
#pragma unroll
      for (int q = 0; q < 8; ++q) {
        int m = mBase + mi * 16 + sub * 8 + q;
        size_t oi = (size_t)m * OOd + n;
        float v = gv[q] * (acc[mi][ni][q] + bias);
        outp[oi] = accumulate ? (outp[oi] + v) : v;
      }
    }
  }
}

// ---------------- unpack: tmp [A,O] f32 -> out [B,O,N,P] (LDS tile transpose) --
__global__ __launch_bounds__(256) void unpack_out(const float* __restrict__ tmp,
                                                  float* __restrict__ out) {
  __shared__ float tile[32][33];
  int rem0 = blockIdx.x * 32;
  int o0 = blockIdx.y * 32;
  int b = blockIdx.z;
  int tx = threadIdx.x, ty = threadIdx.y;
  for (int i = ty; i < 32; i += 8)
    tile[i][tx] = tmp[(size_t)(b * NPP + rem0 + i) * OOd + o0 + tx];  // [remi][oi]
  __syncthreads();
  for (int i = ty; i < 32; i += 8)
    out[((size_t)(b * OOd + o0 + i)) * NPP + rem0 + tx] = tile[tx][i];
}

// ---------------- host launch ----------------
extern "C" void kernel_launch(void* const* d_in, const int* in_sizes, int n_in,
                              void* d_out, int out_size, void* d_ws, size_t ws_size,
                              hipStream_t stream) {
  (void)in_sizes; (void)n_in; (void)out_size; (void)ws_size;
  const float* x  = (const float*)d_in[0];
  const float* Wg = (const float*)d_in[1];
  const float* bg = (const float*)d_in[2];
  const float* Wn = (const float*)d_in[3];
  const float* bn = (const float*)d_in[4];
  const float* W1 = (const float*)d_in[5];
  const float* b1 = (const float*)d_in[6];
  const float* W2 = (const float*)d_in[7];
  const float* b2 = (const float*)d_in[8];
  const float* nu = (const float*)d_in[9];

  char* ws = (char*)d_ws;
  size_t off = 0;
  auto take = [&](size_t bytes) -> char* {
    char* p = ws + off;
    off += (bytes + 255) & ~(size_t)255;
    return p;
  };
  _Float16* xh    = (_Float16*)take((size_t)AA * CC * 2);        //  32 MB
  _Float16* w1t   = (_Float16*)take((size_t)EE * HHd * CC * 2);  //  24 MB
  _Float16* w2t   = (_Float16*)take((size_t)EE * OOd * HHd * 2); //  24 MB
  _Float16* hh    = (_Float16*)take((size_t)AA * HHd * 2);       // 128 MB (L2-resident)
  float*    gates = (float*)take((size_t)AA * EE * 4);
  float*    tmp   = (float*)take((size_t)AA * OOd * 4);          //  64 MB

  router_kernel<<<AA / 256, 256, 0, stream>>>(x, Wg, bg, Wn, bn, nu, gates);
  pack_x_kernel<<<dim3(AA / 32, CC / 32), dim3(32, 8), 0, stream>>>(x, xh);
  transpose_f32_f16<<<dim3(HHd / 32, CC / 32, EE), dim3(32, 8), 0, stream>>>(W1, w1t, CC, HHd);
  transpose_f32_f16<<<dim3(OOd / 32, HHd / 32, EE), dim3(32, 8), 0, stream>>>(W2, w2t, HHd, OOd);

  for (int e = 0; e < EE; ++e) {
    gemm1_gelu<<<dim3(AA / 128, HHd / 256), 256, 0, stream>>>(
        xh, w1t + (size_t)e * HHd * CC, b1 + (size_t)e * HHd, hh);
    gemm2_gate<<<dim3(AA / 128, OOd / 256), 256, 0, stream>>>(
        hh, w2t + (size_t)e * OOd * HHd, b2 + (size_t)e * OOd, gates, e, e > 0 ? 1 : 0, tmp);
  }
  unpack_out<<<dim3(NPP / 32, OOd / 32, BB), dim3(32, 8), 0, stream>>>(tmp, (float*)d_out);
}
